// TransformerLayer_WithCLSToken_88768384073824
// MI455X (gfx1250) — compile-verified, matrix operand order checked by reading, not run on hardware
//
#include <hip/hip_runtime.h>

// ---------------------------------------------------------------- constants
#define Eq   512
#define Hh   8
#define DHd  64
#define Ff   2048
#define KSn  33
#define NCLS 1
#define Bb   2
#define Ll   4096
#define Tt   (Bb * (Ll + NCLS))      // 8194 tokens
#define EPSf 1e-5f

typedef __bf16 bf16;
typedef __attribute__((ext_vector_type(16))) __bf16 v16bf;
typedef __attribute__((ext_vector_type(8)))  float  v8f;
typedef int v4i_vs __attribute__((vector_size(16)));   // builtin's int4 type

union Frag16 { v16bf v; uint4 u[2]; };

// ---- CDNA5 async global->LDS copy (ASYNCcnt-tracked, bypasses VGPRs)
// probe-confirmed signature: (AS1 int4* src, AS3 int4* dst, imm offset, imm cpol)
#if __has_builtin(__builtin_amdgcn_global_load_async_to_lds_b128)
#define ASYNC_LDS_COPY_B128(ldsp, gp)                                         \
  __builtin_amdgcn_global_load_async_to_lds_b128(                             \
      (__attribute__((address_space(1))) v4i_vs*)(unsigned long long)(gp),    \
      (__attribute__((address_space(3))) v4i_vs*)(unsigned)(unsigned long long)(ldsp), \
      0, 0)
#else
#define ASYNC_LDS_COPY_B128(ldsp, gp)                                        \
  asm volatile("global_load_async_to_lds_b128 %0, %1, off"                   \
               :: "v"((unsigned)(unsigned long long)(ldsp)),                 \
                  "v"((const void*)(gp))                                     \
               : "memory")
#endif

#if __has_builtin(__builtin_amdgcn_s_wait_asynccnt)
#define WAIT_ASYNC() __builtin_amdgcn_s_wait_asynccnt(0)
#else
#define WAIT_ASYNC() asm volatile("s_wait_asynccnt 0x0" ::: "memory")
#endif

// ---------------------------------------------------------------- helpers
__device__ __forceinline__ float wave_sum32(float p) {
#pragma unroll
  for (int o = 16; o > 0; o >>= 1) p += __shfl_xor(p, o, 32);
  return p;
}

// ---------------------------------------------------------------- f32 -> bf16
__global__ void __launch_bounds__(256)
cvt_bf16_kernel(const float* __restrict__ src, bf16* __restrict__ dst, int n) {
  int i = blockIdx.x * 256 + threadIdx.x;
  if (i < n) dst[i] = (bf16)src[i];
}

// ---------------------------------------------------------------- LayerNorm
// mode 0: write bf16 to out_main[t*E]
// mode 1: split — local tokens packed to out_main[(b*L+l)*E], CLS to out_cls[b*E]
__global__ void __launch_bounds__(256)
layernorm_kernel(const float* __restrict__ x, const float* __restrict__ g,
                 const float* __restrict__ bv, bf16* __restrict__ out_main,
                 bf16* __restrict__ out_cls, int mode) {
  const int t    = blockIdx.x;          // token id, 0..T-1
  const int tid  = threadIdx.x;         // 0..255, each owns 2 of E=512
  const int lane = tid & 31, wave = tid >> 5;
  const float* xp = x + (size_t)t * Eq;
  float v0 = xp[tid], v1 = xp[tid + 256];

  __shared__ float red[8];
  float s = wave_sum32(v0 + v1);
  if (lane == 0) red[wave] = s;
  __syncthreads();
  float mu = 0.f;
#pragma unroll
  for (int i = 0; i < 8; ++i) mu += red[i];
  mu *= (1.0f / Eq);
  __syncthreads();

  float d0 = v0 - mu, d1 = v1 - mu;
  float q = wave_sum32(d0 * d0 + d1 * d1);
  if (lane == 0) red[wave] = q;
  __syncthreads();
  float var = 0.f;
#pragma unroll
  for (int i = 0; i < 8; ++i) var += red[i];
  var *= (1.0f / Eq);
  float rstd = rsqrtf(var + EPSf);

  bf16 o0 = (bf16)(d0 * rstd * g[tid]       + bv[tid]);
  bf16 o1 = (bf16)(d1 * rstd * g[tid + 256] + bv[tid + 256]);

  bf16* dst;
  if (mode == 0) {
    dst = out_main + (size_t)t * Eq;
  } else {
    int b = t / (Ll + NCLS), i = t % (Ll + NCLS);
    dst = (i < Ll) ? out_main + ((size_t)b * Ll + i) * Eq
                   : out_cls  + (size_t)b * Eq;
  }
  dst[tid] = o0; dst[tid + 256] = o1;
}

// ---------------------------------------------------------------- WMMA GEMM
// C[M,N] = act( A[M,K](bf16) @ W[N,K]^T(bf16) + bias ) (+ resid).
// Block tile 128x128, 8 waves: 2 (M) x 4 (N), wave tile 64x32.
// K staged 32/iter through double-buffered LDS via async global->LDS copies.
#define TM 128
#define TN 128
#define TK 32
#define LDSK (TK + 8)     // 80B row stride, 16B aligned

__global__ void __launch_bounds__(256)
wmma_gemm_kernel(const bf16* __restrict__ A, int lda,
                 const bf16* __restrict__ W, int ldw,
                 const float* __restrict__ bias,
                 const float* __restrict__ resid, int ldr,
                 float* __restrict__ outf, int ldo,
                 bf16* __restrict__ outb, int ldob,
                 int M, int N, int K, int silu) {
  __shared__ bf16 As[2][TM][LDSK];
  __shared__ bf16 Ws[2][TN][LDSK];

  const int tid  = threadIdx.x;
  const int lane = tid & 31;
  const int wave = tid >> 5;
  const int wm   = wave & 1;        // 2 waves along M (64 rows each)
  const int wn   = wave >> 1;       // 4 waves along N (32 cols each)
  const int bm   = blockIdx.y * TM;
  const int bn   = blockIdx.x * TN;
  const int lh   = lane >> 4;       // K half (0/1)
  const int lm   = lane & 15;

  // staging: 2 threads per row, each covers 32B (2 x b128) of A and of W
  const int srow = tid >> 1;               // 0..127
  const int scol = (tid & 1) * 16;         // elem offset 0 / 16

  const bf16* Ag = A + (size_t)(bm + srow) * lda + scol;
  const bf16* Wg = W + (size_t)(bn + srow) * ldw + scol;

  v8f acc[4][2] = {};

  const int nk = K / TK;
  // prologue: fill buffer 0
  {
    ASYNC_LDS_COPY_B128(&As[0][srow][scol],     Ag);
    ASYNC_LDS_COPY_B128(&As[0][srow][scol + 8], Ag + 8);
    ASYNC_LDS_COPY_B128(&Ws[0][srow][scol],     Wg);
    ASYNC_LDS_COPY_B128(&Ws[0][srow][scol + 8], Wg + 8);
  }

  for (int kt = 0; kt < nk; ++kt) {
    const int buf = kt & 1;
    WAIT_ASYNC();
    __syncthreads();

    if (kt + 1 < nk) {   // prefetch next K tile into the other buffer
      const bf16* an  = Ag + (size_t)(kt + 1) * TK;
      const bf16* wnx = Wg + (size_t)(kt + 1) * TK;
      ASYNC_LDS_COPY_B128(&As[buf ^ 1][srow][scol],     an);
      ASYNC_LDS_COPY_B128(&As[buf ^ 1][srow][scol + 8], an + 8);
      ASYNC_LDS_COPY_B128(&Ws[buf ^ 1][srow][scol],     wnx);
      ASYNC_LDS_COPY_B128(&Ws[buf ^ 1][srow][scol + 8], wnx + 8);
    }

    Frag16 af[4], wf[2];
#pragma unroll
    for (int mi = 0; mi < 4; ++mi) {
      const bf16* s = &As[buf][wm * 64 + mi * 16 + lm][lh * 16];
      af[mi].u[0] = *(const uint4*)s;
      af[mi].u[1] = *(const uint4*)(s + 8);
    }
#pragma unroll
    for (int ni = 0; ni < 2; ++ni) {
      const bf16* s = &Ws[buf][wn * 32 + ni * 16 + lm][lh * 16];
      wf[ni].u[0] = *(const uint4*)s;
      wf[ni].u[1] = *(const uint4*)(s + 8);
    }

#pragma unroll
    for (int mi = 0; mi < 4; ++mi)
#pragma unroll
      for (int ni = 0; ni < 2; ++ni)
        acc[mi][ni] = __builtin_amdgcn_wmma_f32_16x16x32_bf16(
            false, af[mi].v, false, wf[ni].v, (short)0, acc[mi][ni],
            false, false);
  }

  // epilogue: D VGPR j holds rows {j, j+8}; lane halves split M
  const int baseRow = bm + wm * 64;
  const int baseCol = bn + wn * 32;
#pragma unroll
  for (int mi = 0; mi < 4; ++mi) {
#pragma unroll
    for (int ni = 0; ni < 2; ++ni) {
      const int col = baseCol + ni * 16 + lm;
#pragma unroll
      for (int j = 0; j < 8; ++j) {
        const int row = baseRow + mi * 16 + j + lh * 8;
        if (row < M) {
          float v = acc[mi][ni][j] + bias[col];
          if (silu) v = v / (1.0f + __expf(-v));
          if (resid) v += resid[(size_t)row * ldr + col];
          if (outf) outf[(size_t)row * ldo + col] = v;
          if (outb) outb[(size_t)row * ldob + col] = (bf16)v;
        }
      }
    }
  }
}

// ---------------------------------------------------------------- NATTEN core
// one wave per (b,l,h); lane owns dims {lane, lane+32}; online softmax
__global__ void __launch_bounds__(256)
natten_kernel(const float* __restrict__ qkv, bf16* __restrict__ attn_out) {
  const int gw   = blockIdx.x * 8 + (threadIdx.x >> 5);
  const int lane = threadIdx.x & 31;
  const int h = gw % Hh, l = (gw / Hh) % Ll, b = gw / (Hh * Ll);

  int start = l - KSn / 2;
  if (start < 0) start = 0;
  if (start > Ll - KSn) start = Ll - KSn;

  const float scale = 0.125f;   // DH^-0.5
  const size_t qoff = ((size_t)(b * Ll + l)) * (3 * Eq) + h * DHd;
  const float q0 = qkv[qoff + lane]      * scale;
  const float q1 = qkv[qoff + lane + 32] * scale;

  float m = -1e30f, ssum = 0.f, acc0 = 0.f, acc1 = 0.f;
  for (int j = 0; j < KSn; ++j) {
    const size_t roff = ((size_t)(b * Ll + start + j)) * (3 * Eq) + h * DHd;
    float p = wave_sum32(q0 * qkv[roff + Eq + lane] +
                         q1 * qkv[roff + Eq + lane + 32]);
    float mn = fmaxf(m, p);
    float corr = __expf(m - mn);
    float w    = __expf(p - mn);
    ssum = ssum * corr + w;
    acc0 = acc0 * corr + w * qkv[roff + 2 * Eq + lane];
    acc1 = acc1 * corr + w * qkv[roff + 2 * Eq + lane + 32];
    m = mn;
  }
  const float inv = 1.0f / ssum;
  const size_t o = ((size_t)(b * Ll + l)) * Eq + h * DHd;
  attn_out[o + lane]      = (bf16)(acc0 * inv);
  attn_out[o + lane + 32] = (bf16)(acc1 * inv);
}

// ---------------------------------------------------------------- CLS MHA
// one wave per (b,h); q already scaled+biased; online softmax over L+1 keys
__global__ void __launch_bounds__(32)
mha_cls_kernel(const float* __restrict__ qcls, const float* __restrict__ kg,
               const float* __restrict__ vg, bf16* __restrict__ mha_out) {
  const int b = blockIdx.x / Hh, h = blockIdx.x % Hh;
  const int lane = threadIdx.x;
  const float q0 = qcls[(size_t)b * Eq + h * DHd + lane];
  const float q1 = qcls[(size_t)b * Eq + h * DHd + lane + 32];

  float m = -1e30f, ssum = 0.f, acc0 = 0.f, acc1 = 0.f;
  for (int i = 0; i < Ll + NCLS; ++i) {
    const size_t off = ((size_t)(b * (Ll + NCLS) + i)) * Eq + h * DHd;
    float p = wave_sum32(q0 * kg[off + lane] + q1 * kg[off + lane + 32]);
    float mn = fmaxf(m, p);
    float corr = __expf(m - mn);
    float w    = __expf(p - mn);
    ssum = ssum * corr + w;
    acc0 = acc0 * corr + w * vg[off + lane];
    acc1 = acc1 * corr + w * vg[off + lane + 32];
    m = mn;
  }
  const float inv = 1.0f / ssum;
  mha_out[(size_t)b * Eq + h * DHd + lane]      = (bf16)(acc0 * inv);
  mha_out[(size_t)b * Eq + h * DHd + lane + 32] = (bf16)(acc1 * inv);
}

// ---------------------------------------------------------------- tiny GEMM
// out[m,n] = alpha*(A[m,:]·W[n,:] + bias[n]) (+resid); M is tiny (CLS rows)
__global__ void __launch_bounds__(256)
small_gemm_kernel(const bf16* __restrict__ A, int lda,
                  const bf16* __restrict__ W, int ldw,
                  const float* __restrict__ bias, float alpha,
                  const float* __restrict__ resid, int ldr,
                  float* __restrict__ outf, int ldo,
                  int Mrows, int N, int K) {
  int idx = blockIdx.x * 256 + threadIdx.x;
  if (idx >= Mrows * N) return;
  int m = idx / N, n = idx % N;
  float s = 0.f;
  for (int k = 0; k < K; ++k)
    s += (float)A[(size_t)m * lda + k] * (float)W[(size_t)n * ldw + k];
  float v = alpha * (s + bias[n]);
  if (resid) v += resid[(size_t)m * ldr + n];
  outf[(size_t)m * ldo + n] = v;
}

// ---------------------------------------------------------------- glue
__global__ void __launch_bounds__(256)
build_tmpkey_kernel(const bf16* __restrict__ xloc, const bf16* __restrict__ xcls,
                    bf16* __restrict__ tmpk) {
  int i = blockIdx.x * 256 + threadIdx.x;
  if (i >= Tt * Eq) return;
  int t = i / Eq, e = i % Eq;
  int b = t / (Ll + NCLS), r = t % (Ll + NCLS);
  tmpk[i] = (r < Ll) ? xloc[((size_t)b * Ll + r) * Eq + e]
                     : xcls[(size_t)b * Eq + e];
}

__global__ void __launch_bounds__(256)
resid_local_kernel(const float* __restrict__ x, const float* __restrict__ xloc,
                   float* __restrict__ x1) {
  int i = blockIdx.x * 256 + threadIdx.x;
  if (i >= Bb * Ll * Eq) return;
  int e = i % Eq, l = (i / Eq) % Ll, b = i / (Eq * Ll);
  size_t tok = (size_t)b * (Ll + NCLS) + l;
  x1[tok * Eq + e] = x[tok * Eq + e] + xloc[i];
}

// ---------------------------------------------------------------- launch
extern "C" void kernel_launch(void* const* d_in, const int* in_sizes, int n_in,
                              void* d_out, int out_size, void* d_ws, size_t ws_size,
                              hipStream_t stream) {
  const float* x        = (const float*)d_in[0];
  const float* ln1_g    = (const float*)d_in[1];
  const float* ln1_b    = (const float*)d_in[2];
  const float* qkv_w    = (const float*)d_in[3];
  const float* qkv_b    = (const float*)d_in[4];
  const float* naproj_w = (const float*)d_in[5];
  const float* naproj_b = (const float*)d_in[6];
  const float* inproj_w = (const float*)d_in[7];
  const float* inproj_b = (const float*)d_in[8];
  const float* out_w    = (const float*)d_in[9];
  const float* out_b    = (const float*)d_in[10];
  const float* ln2_g    = (const float*)d_in[11];
  const float* ln2_b    = (const float*)d_in[12];
  const float* fc1_w    = (const float*)d_in[13];
  const float* fc1_b    = (const float*)d_in[14];
  const float* fc2_w    = (const float*)d_in[15];
  const float* fc2_b    = (const float*)d_in[16];

  // ---- workspace carve (256B aligned)
  char* p = (char*)d_ws;
  auto carve = [&](size_t bytes) -> void* {
    void* r = (void*)p;
    p += (bytes + 255) & ~(size_t)255;
    return r;
  };
  const int Tpad = 8320;  // T rounded up to multiple of 128 (GEMM A padding)

  bf16* wqkv  = (bf16*)carve((size_t)3 * Eq * Eq * 2);
  bf16* wnap  = (bf16*)carve((size_t)Eq * Eq * 2);
  bf16* winp  = (bf16*)carve((size_t)3 * Eq * Eq * 2);
  bf16* wout  = (bf16*)carve((size_t)Eq * Eq * 2);
  bf16* wfc1  = (bf16*)carve((size_t)Ff * Eq * 2);
  bf16* wfc2  = (bf16*)carve((size_t)Eq * Ff * 2);

  bf16*  xn_na  = (bf16*)carve((size_t)Bb * Ll * Eq * 2);    // packed local xn
  bf16*  xn_cls = (bf16*)carve((size_t)Bb * Eq * 2);
  float* qkv    = (float*)carve((size_t)Bb * Ll * 3 * Eq * 4);
  bf16*  attn   = (bf16*)carve((size_t)Bb * Ll * Eq * 2);    // natten pre-proj
  float* xloc_f = (float*)carve((size_t)Bb * Ll * Eq * 4);   // natten_1d output
  bf16*  xloc_b = (bf16*)carve((size_t)Bb * Ll * Eq * 2);
  bf16*  tmpk   = (bf16*)carve((size_t)Tpad * Eq * 2);       // concat key input
  float* kg     = (float*)carve((size_t)Tpad * Eq * 4);
  float* vg     = (float*)carve((size_t)Tpad * Eq * 4);
  float* qcls   = (float*)carve((size_t)Bb * Eq * 4);
  bf16*  mhab   = (bf16*)carve((size_t)Bb * Eq * 2);
  float* x1     = (float*)carve((size_t)Tt * Eq * 4);        // first residual
  bf16*  xn2    = (bf16*)carve((size_t)Tpad * Eq * 2);
  bf16*  hbuf   = (bf16*)carve((size_t)Tpad * Ff * 2);

  // ---- 1. weights -> bf16
  auto cvt = [&](const float* s, bf16* d, int n) {
    cvt_bf16_kernel<<<(n + 255) / 256, 256, 0, stream>>>(s, d, n);
  };
  cvt(qkv_w,    wqkv, 3 * Eq * Eq);
  cvt(naproj_w, wnap, Eq * Eq);
  cvt(inproj_w, winp, 3 * Eq * Eq);
  cvt(out_w,    wout, Eq * Eq);
  cvt(fc1_w,    wfc1, Ff * Eq);
  cvt(fc2_w,    wfc2, Eq * Ff);

  // ---- 2. LN1 (split local / CLS)
  layernorm_kernel<<<Tt, 256, 0, stream>>>(x, ln1_g, ln1_b, xn_na, xn_cls, 1);

  // ---- 3. qkv GEMM: [8192,512] x [1536,512]^T
  wmma_gemm_kernel<<<dim3(3 * Eq / TN, (Bb * Ll) / TM), 256, 0, stream>>>(
      xn_na, Eq, wqkv, Eq, qkv_b, nullptr, 0, qkv, 3 * Eq, nullptr, 0,
      Bb * Ll, 3 * Eq, Eq, 0);

  // ---- 4. neighborhood attention
  natten_kernel<<<(Bb * Ll * Hh) / 8, 256, 0, stream>>>(qkv, attn);

  // ---- 5. na_proj GEMM (fp32 + bf16 out)
  wmma_gemm_kernel<<<dim3(Eq / TN, (Bb * Ll) / TM), 256, 0, stream>>>(
      attn, Eq, wnap, Eq, naproj_b, nullptr, 0, xloc_f, Eq, xloc_b, Eq,
      Bb * Ll, Eq, Eq, 0);

  // ---- 6. tmp_key = concat(x_local, xn_cls)
  build_tmpkey_kernel<<<(Tt * Eq + 255) / 256, 256, 0, stream>>>(xloc_b, xn_cls, tmpk);

  // ---- 7/8. k,v projections for global MHA: M=8194
  wmma_gemm_kernel<<<dim3(Eq / TN, (Tt + TM - 1) / TM), 256, 0, stream>>>(
      tmpk, Eq, winp + (size_t)Eq * Eq, Eq, inproj_b + Eq, nullptr, 0,
      kg, Eq, nullptr, 0, Tt, Eq, Eq, 0);
  wmma_gemm_kernel<<<dim3(Eq / TN, (Tt + TM - 1) / TM), 256, 0, stream>>>(
      tmpk, Eq, winp + (size_t)2 * Eq * Eq, Eq, inproj_b + 2 * Eq, nullptr, 0,
      vg, Eq, nullptr, 0, Tt, Eq, Eq, 0);

  // ---- 9. q projection for CLS (M=2), scale folded in
  small_gemm_kernel<<<(Bb * Eq + 255) / 256, 256, 0, stream>>>(
      xn_cls, Eq, winp, Eq, inproj_b, 0.125f, nullptr, 0, qcls, Eq,
      Bb, Eq, Eq);

  // ---- 10. CLS attention over L+1 keys
  mha_cls_kernel<<<Bb * Hh, 32, 0, stream>>>(qcls, kg, vg, mhab);

  // ---- 11. CLS out-proj + residual, written into x1 CLS rows
  small_gemm_kernel<<<(Bb * Eq + 255) / 256, 256, 0, stream>>>(
      mhab, Eq, wout, Eq, out_b, 1.0f,
      x + (size_t)Ll * Eq, (Ll + NCLS) * Eq,
      x1 + (size_t)Ll * Eq, (Ll + NCLS) * Eq,
      Bb, Eq, Eq);

  // ---- 12. residual for local rows
  resid_local_kernel<<<(Bb * Ll * Eq + 255) / 256, 256, 0, stream>>>(x, xloc_f, x1);

  // ---- 13. LN2
  layernorm_kernel<<<Tt, 256, 0, stream>>>(x1, ln2_g, ln2_b, xn2, nullptr, 0);

  // ---- 14. fc1 + SiLU -> bf16 h
  wmma_gemm_kernel<<<dim3(Ff / TN, (Tt + TM - 1) / TM), 256, 0, stream>>>(
      xn2, Eq, wfc1, Eq, fc1_b, nullptr, 0, nullptr, 0, hbuf, Ff,
      Tt, Ff, Eq, 1);

  // ---- 15. fc2 + residual -> d_out (fp32)
  wmma_gemm_kernel<<<dim3(Eq / TN, (Tt + TM - 1) / TM), 256, 0, stream>>>(
      hbuf, Ff, wfc2, Ff, fc2_b, x1, Eq, (float*)d_out, Eq, nullptr, 0,
      Tt, Eq, Ff, 0);

  (void)in_sizes; (void)n_in; (void)out_size; (void)ws_size;
}
